// Where2comm_46377056862294
// MI455X (gfx1250) — matrix operand order repaired and checked by reading, not compiled, not run on hardware
//
#include <hip/hip_runtime.h>
#include <hip/hip_bf16.h>

typedef __attribute__((ext_vector_type(16))) _Float16 v16h;
typedef __attribute__((ext_vector_type(8)))  _Float16 v8h;
typedef __attribute__((ext_vector_type(8)))  float    v8f;
typedef __attribute__((ext_vector_type(8)))  int      v8i;

#define BB 2
#define LL 5
#define CC 256
#define HH 100
#define WW 252
#define HWW (HH * WW)
#define NN (BB * LL)
#define TILES_PER_B (HWW / 16)          /* 1575 */
#define TOTAL_TILES (BB * TILES_PER_B)  /* 3150 */
#define CPAD (CC + 8)                   /* row pad: lanes land on banks 4 apart */

// 1/(2*pi*sigma) * exp(-(dx^2+dy^2)/(2*sigma^2)), sigma = 1
__device__ __constant__ float GWK[5][5] = {
  {0.00291502f, 0.01306423f, 0.02153928f, 0.01306423f, 0.00291502f},
  {0.01306423f, 0.05854983f, 0.09653235f, 0.05854983f, 0.01306423f},
  {0.02153928f, 0.09653235f, 0.15915494f, 0.09653235f, 0.02153928f},
  {0.01306423f, 0.05854983f, 0.09653235f, 0.05854983f, 0.01306423f},
  {0.00291502f, 0.01306423f, 0.02153928f, 0.01306423f, 0.00291502f}
};

// conf = max over the A=2 channels of sigmoid(psm)
__global__ void conf_kernel(const float* __restrict__ psm, float* __restrict__ conf) {
  int idx = blockIdx.x * blockDim.x + threadIdx.x;
  if (idx >= NN * HWW) return;
  int n = idx / HWW, p = idx % HWW;
  const float* base = psm + (size_t)n * 2 * HWW + p;
  float s0 = 1.0f / (1.0f + __expf(-base[0]));
  float s1 = 1.0f / (1.0f + __expf(-base[HWW]));
  conf[idx] = fmaxf(s0, s1);
}

// 5x5 gaussian blur (zero pad) -> threshold -> force-on even agents
__global__ void mask_kernel(const float* __restrict__ conf, float* __restrict__ mask) {
  int idx = blockIdx.x * blockDim.x + threadIdx.x;
  if (idx >= NN * HWW) return;
  int n = idx / HWW, pix = idx % HWW;
  int h = pix / WW, w = pix % WW;
  const float* cimg = conf + (size_t)n * HWW;
  float s = 0.0f;
  #pragma unroll
  for (int dy = -2; dy <= 2; ++dy) {
    int y = h + dy;
    if (y < 0 || y >= HH) continue;
    #pragma unroll
    for (int dx = -2; dx <= 2; ++dx) {
      int xw = w + dx;
      if (xw < 0 || xw >= WW) continue;
      s += GWK[dy + 2][dx + 2] * cimg[y * WW + xw];
    }
  }
  float m = (s > 0.01f) ? 1.0f : 0.0f;
  if (((n % LL) & 1) == 0) m = 1.0f;   // mask[:, ::2] = 1
  mask[idx] = m;
}

// One wave32 workgroup per 16-pixel tile:
//  - bilinear warp (comm mask + validity folded into corner weights), gathered
//    once per (pixel, agent, channel) and stashed as f16 in LDS
//  - per-pixel 5-agent attention scores via v_wmma_f32_16x16x32_f16
//    (diagonal trick; ego A-fragment is bf0 itself, same K order as B)
//  - softmax-weighted feature sum read back from LDS (no global re-gather)
__global__ __launch_bounds__(32)
void fuse_kernel(const float* __restrict__ xin, const float* __restrict__ tmat,
                 const float* __restrict__ maskp, float* __restrict__ out) {
  __shared__ __align__(16) _Float16 sfeat[LL][16][CPAD];   // ~41.3 KB

  const int lane = threadIdx.x & 31;
  const int wid  = blockIdx.x;
  if (wid >= TOTAL_TILES) return;               // wave-uniform: EXEC stays all-ones
  const int b    = wid / TILES_PER_B;
  const int pix0 = (wid % TILES_PER_B) * 16;
  const int p    = lane & 15;                   // pixel slot == WMMA row/col
  const int hi   = lane >> 4;                   // lane half (K-range selector)
  const int mypix = pix0 + p;
  const int hq = mypix / WW, wq = mypix % WW;
  const int hi16 = hi * 16;

  // ---- per-agent bilinear corners (validity + comm mask folded into weights)
  int   off[LL][4];
  float wt [LL][4];
  const float xs = (wq + 0.5f) * (2.0f / WW) - 1.0f;
  const float ys = (hq + 0.5f) * (2.0f / HH) - 1.0f;
  #pragma unroll
  for (int l = 0; l < LL; ++l) {
    const float* T = tmat + ((size_t)(b * LL + 0) * LL + l) * 16;  // t[b,0,l] (4x4)
    float t00 = T[0];
    float t01 = T[1] * ((float)HH / (float)WW);
    float t02 = T[3] * (2.0f / (4.0f * 0.4f * (float)WW));
    float t10 = T[4] * ((float)WW / (float)HH);
    float t11 = T[5];
    float t12 = T[7] * (2.0f / (4.0f * 0.4f * (float)HH));
    float gx = xs * t00 + ys * t01 + t02;
    float gy = xs * t10 + ys * t11 + t12;
    float sx = ((gx + 1.0f) * WW - 1.0f) * 0.5f;
    float sy = ((gy + 1.0f) * HH - 1.0f) * 0.5f;
    float x0f = floorf(sx), y0f = floorf(sy);
    int   x0 = (int)x0f,   y0 = (int)y0f;
    float fx = sx - x0f,   fy = sy - y0f;
    const float* mimg = maskp + (size_t)(b * LL + l) * HWW;
    const int   cx[2] = {x0, x0 + 1},       cy[2] = {y0, y0 + 1};
    const float wx[2] = {1.0f - fx, fx},    wy[2] = {1.0f - fy, fy};
    #pragma unroll
    for (int ky = 0; ky < 2; ++ky) {
      #pragma unroll
      for (int kx = 0; kx < 2; ++kx) {
        int ix = cx[kx], iy = cy[ky];
        bool valid = (ix >= 0) & (ix < WW) & (iy >= 0) & (iy < HH);
        int ixc = min(max(ix, 0), WW - 1);
        int iyc = min(max(iy, 0), HH - 1);
        int o = iyc * WW + ixc;
        off[l][ky * 2 + kx] = o;
        wt [l][ky * 2 + kx] = valid ? (wx[kx] * wy[ky] * mimg[o]) : 0.0f;
      }
    }
  }

  auto gather = [&](int l, int c) -> float {
    const float* img = xin + ((size_t)(b * LL + l) * CC + c) * HWW;
    return wt[l][0] * img[off[l][0]] + wt[l][1] * img[off[l][1]]
         + wt[l][2] * img[off[l][2]] + wt[l][3] * img[off[l][3]];
  };
  auto stash = [&](int l, int cb, v16h v) {     // 2x ds_store_b128 per fragment
    _Float16* rp = &sfeat[l][p][cb + hi16];
    *(v8h*)(rp)     = __builtin_shufflevector(v, v, 0, 1, 2, 3, 4, 5, 6, 7);
    *(v8h*)(rp + 8) = __builtin_shufflevector(v, v, 8, 9, 10, 11, 12, 13, 14, 15);
  };

  // ---- phase A: gather + stage to LDS + WMMA scores (diagonal = per-pixel dot)
  v8f acc[LL] = {};
  for (int k = 0; k < CC / 32; ++k) {
    const int cb = k * 32;

    // agent-0 (ego) B-fragment: column N = pixel (lane&15), K = hi16+i
    v16h bf0;
    #pragma unroll
    for (int i = 0; i < 16; ++i) bf0[i] = (_Float16)gather(0, cb + hi16 + i);
    stash(0, cb, bf0);

    // ego A-fragment: row M = pixel (lane&15), same K = hi16+i order, copied from bf0
    v8i bi = __builtin_bit_cast(v8i, bf0), pi;
    #pragma unroll
    for (int d = 0; d < 8; ++d) pi[d] = __shfl_xor(bi[d], 16, 32);
    v16h po = __builtin_bit_cast(v16h, pi);
    v16h a;
    #pragma unroll
    for (int j = 0; j < 8; ++j) {
      a[j]     = bf0[j];                        // K = hi*16 + j
      a[j + 8] = bf0[j + 8];                    // K = hi*16 + 8 + j
    }

    acc[0] = __builtin_amdgcn_wmma_f32_16x16x32_f16(
                 false, a, false, bf0, (short)0, acc[0], false, false);
    #pragma unroll
    for (int l = 1; l < LL; ++l) {
      v16h bf;
      #pragma unroll
      for (int i = 0; i < 16; ++i) bf[i] = (_Float16)gather(l, cb + hi16 + i);
      stash(l, cb, bf);
      acc[l] = __builtin_amdgcn_wmma_f32_16x16x32_f16(
                   false, a, false, bf, (short)0, acc[l], false, false);
    }
  }

  // ---- diagonal extraction + softmax over agents (per pixel)
  float attn[LL];
  {
    const bool mine = (hi == (p >> 3));  // this half-lane holds row p of col p
    const int  ridx = p & 7;
    float sc[LL];
    #pragma unroll
    for (int l = 0; l < LL; ++l) {
      float t = 0.0f;
      #pragma unroll
      for (int r = 0; r < 8; ++r) if (r == ridx) t = acc[l][r];
      t = mine ? t : 0.0f;
      t += __shfl_xor(t, 16, 32);        // combine the two half-lanes of pixel p
      sc[l] = t * 0.0625f;               // 1/sqrt(C), C=256
    }
    float mx = sc[0];
    #pragma unroll
    for (int l = 1; l < LL; ++l) mx = fmaxf(mx, sc[l]);
    float s = 0.0f;
    #pragma unroll
    for (int l = 0; l < LL; ++l) { attn[l] = __expf(sc[l] - mx); s += attn[l]; }
    float inv = 1.0f / s;
    #pragma unroll
    for (int l = 0; l < LL; ++l) attn[l] *= inv;
  }

  __syncthreads();   // single-wave workgroup: lowers to S_NOP, orders LDS for compiler

  // ---- phase B: attn-weighted sum straight from LDS, coalesced 16-pixel stores
  float* outp = out + (size_t)b * CC * HWW + mypix;
  for (int i = 0; i < CC / 2; ++i) {
    const int c = 2 * i + hi;
    float v = 0.0f;
    #pragma unroll
    for (int l = 0; l < LL; ++l) v += attn[l] * (float)sfeat[l][p][c];
    outp[(size_t)c * HWW] = v;
  }
}

extern "C" void kernel_launch(void* const* d_in, const int* in_sizes, int n_in,
                              void* d_out, int out_size, void* d_ws, size_t ws_size,
                              hipStream_t stream) {
  const float* x    = (const float*)d_in[0];
  const float* psm  = (const float*)d_in[1];
  const float* tmat = (const float*)d_in[2];
  float* out  = (float*)d_out;
  float* conf = (float*)d_ws;                        // NN*HWW floats
  float* mask = conf + (size_t)NN * HWW;             // NN*HWW floats

  const int np = NN * HWW;
  conf_kernel<<<(np + 255) / 256, 256, 0, stream>>>(psm, conf);
  mask_kernel<<<(np + 255) / 256, 256, 0, stream>>>(conf, mask);

  fuse_kernel<<<TOTAL_TILES, 32, 0, stream>>>(x, tmat, mask, out);
}
